// PointSetPooling_88888643158315
// MI455X (gfx1250) — compile-verified, matrix operand order checked by reading, not run on hardware
//
#include <hip/hip_runtime.h>
#include <hip/hip_bf16.h>

typedef __attribute__((ext_vector_type(16))) _Float16 v16h;
typedef __attribute__((ext_vector_type(8)))  float    v8f;

#define BN_EPS 1e-5f

// ---------------------------------------------------------------------------
// 1) gather src/dest points, diff, write f16 row padded to 32 cols (zeros)
// ---------------------------------------------------------------------------
__global__ void psp_gather_diff(const float* __restrict__ pc,
                                const int* __restrict__ kpi,
                                const int* __restrict__ si,
                                int S, _Float16* __restrict__ diff) {
    int s = blockIdx.x * blockDim.x + threadIdx.x;
    if (s >= S) return;
    int a = si[2 * s];
    int b = kpi[si[2 * s + 1]];
    v16h lo = {};
    lo[0] = (_Float16)(pc[3 * a + 0] - pc[3 * b + 0]);
    lo[1] = (_Float16)(pc[3 * a + 1] - pc[3 * b + 1]);
    lo[2] = (_Float16)(pc[3 * a + 2] - pc[3 * b + 2]);
    v16h hi = {};
    v16h* o = (v16h*)(diff + (size_t)s * 32);   // rows are 64B aligned
    o[0] = lo;
    o[1] = hi;
}

// ---------------------------------------------------------------------------
// 2) pad + swizzle weights into WMMA B-fragment order:
//    Wsw[(((kt*Nt)+tn)*32 + lane)*16 + h] = W[kt*32 + kOf(lane,h)][tn*16 + (lane&15)]
//    kOf(l,h) = h + (h>=8 ? 8 : 0) + ((l>>4)<<3)   (ISA 7.12.2 f16 B layout)
//    -> GEMM loads each B fragment as one contiguous 32-byte v16h.
// ---------------------------------------------------------------------------
__global__ void psp_pad_weights_swz(const float* __restrict__ W, const float* __restrict__ b,
                                    int Kin, int Nout, int Kp, int Np,
                                    _Float16* __restrict__ Wsw, float* __restrict__ bp) {
    int i = blockIdx.x * blockDim.x + threadIdx.x;
    int total = Kp * Np;
    if (i < total) {
        int h    = i & 15;
        int l    = (i >> 4) & 31;
        int tile = i >> 9;               // kt*Nt + tn   (512 elems per tile)
        int Nt = Np >> 4;
        int kt = tile / Nt, tn = tile % Nt;
        int k = kt * 32 + h + ((h >= 8) ? 8 : 0) + ((l >> 4) << 3);
        int n = tn * 16 + (l & 15);
        Wsw[i] = (k < Kin && n < Nout) ? (_Float16)W[k * Nout + n] : (_Float16)0.f;
    }
    if (i < Np) bp[i] = (i < Nout) ? b[i] : 0.f;
}

// ---------------------------------------------------------------------------
// 3) WMMA GEMM: Out(MxNp,f16) = relu(A(MxKp,f16) @ W + bias)
//    Each wave32: one M-tile x NTS N-tiles (compile-time), shared A fragment,
//    f32 accumulation via v_wmma_f32_16x16x32_f16. Wave index is pushed
//    through readfirstlane so all tile math is scalar and the only branch is
//    a scalar s_cbranch -> EXEC stays all-1s around the WMMAs, no saveexec.
// ---------------------------------------------------------------------------
template <int NTS>
__global__ void psp_wmma_gemm_bias_relu(const _Float16* __restrict__ A,
                                        const _Float16* __restrict__ Wsw,
                                        const float* __restrict__ bias,
                                        _Float16* __restrict__ Out,
                                        int M, int Kp, int Np) {
    const int lane = threadIdx.x & 31;
    const int wave = __builtin_amdgcn_readfirstlane(
        blockIdx.x * (blockDim.x >> 5) + (threadIdx.x >> 5));
    const int Mt = M >> 4, Nt = Np >> 4;
    const int Ng = Nt / NTS;                      // exact by construction
    if (wave >= Mt * Ng) return;                  // scalar branch
    const int tm  = wave / Ng;
    const int tn0 = (wave % Ng) * NTS;
    const int row = tm * 16 + (lane & 15);
    const int khi = (lane >> 4) << 3;             // 0 (lanes 0-15) or 8 (16-31)

    v8f acc[NTS] = {};
    const int Kt = Kp >> 5;
    const _Float16* ap = A + (size_t)row * Kp + khi;
    const _Float16* wp = Wsw + (size_t)tn0 * 512 + lane * 16;
    const size_t wkt = (size_t)Nt * 512;          // Wsw stride per k-step
    for (int kt = 0; kt < Kt; ++kt) {
        // A fragment: halves {khi+0..7, khi+16..23} -> two b128 loads
        v16h a;
        #pragma unroll
        for (int h = 0; h < 8; ++h) { a[h] = ap[h]; a[h + 8] = ap[16 + h]; }
        ap += 32;
        #pragma unroll
        for (int t = 0; t < NTS; ++t) {
            const v16h b = *(const v16h*)(wp + (size_t)t * 512);
            acc[t] = __builtin_amdgcn_wmma_f32_16x16x32_f16(
                false, a, false, b, (short)0, acc[t], false, false);
        }
        wp += wkt;
    }
    // C/D layout: VGPR r -> M = tm*16 + r + khi, N = tile_col + (lane&15)
    const int rbase = tm * 16 + khi;
    #pragma unroll
    for (int t = 0; t < NTS; ++t) {
        const int col = (tn0 + t) * 16 + (lane & 15);
        const float bv = bias[col];
        #pragma unroll
        for (int r = 0; r < 8; ++r) {
            float v = acc[t][r] + bv;
            v = v > 0.f ? v : 0.f;                // fused ReLU
            Out[(size_t)(rbase + r) * Np + col] = (_Float16)v;
        }
    }
}

// ---------------------------------------------------------------------------
// 4a) streaming per-column partial sums (coalesced reads; LDS accumulation)
// ---------------------------------------------------------------------------
__global__ void psp_bn_partial(const _Float16* __restrict__ X, size_t total, int ld,
                               float* __restrict__ sum, float* __restrict__ sumsq) {
    __shared__ float ls[320], ls2[320];
    for (int j = threadIdx.x; j < ld; j += blockDim.x) { ls[j] = 0.f; ls2[j] = 0.f; }
    __syncthreads();
    for (size_t i = (size_t)blockIdx.x * blockDim.x + threadIdx.x; i < total;
         i += (size_t)gridDim.x * blockDim.x) {
        float v = (float)X[i];
        int j = (int)(i % ld);
        atomicAdd(&ls[j], v);
        atomicAdd(&ls2[j], v * v);
    }
    __syncthreads();
    for (int j = threadIdx.x; j < ld; j += blockDim.x) {
        if (ls[j] != 0.f) atomicAdd(&sum[j], ls[j]);
        if (ls2[j] != 0.f) atomicAdd(&sumsq[j], ls2[j]);
    }
}

// 4b) finalize: mean / rstd
__global__ void psp_bn_finalize(const float* __restrict__ sum, const float* __restrict__ sumsq,
                                int M, int Nreal, float* __restrict__ mean, float* __restrict__ rstd) {
    int j = blockIdx.x * blockDim.x + threadIdx.x;
    if (j >= Nreal) return;
    float mu = sum[j] / (float)M;
    float var = sumsq[j] / (float)M - mu * mu;
    mean[j] = mu;
    rstd[j] = rsqrtf(var + BN_EPS);
}

// ---------------------------------------------------------------------------
// 5) in-place BN apply on f16 activations (padded cols -> 0)
// ---------------------------------------------------------------------------
__global__ void psp_bn_apply(_Float16* __restrict__ X, size_t total, int ld, int Nreal,
                             const float* __restrict__ mean, const float* __restrict__ rstd,
                             const float* __restrict__ gamma, const float* __restrict__ beta) {
    size_t i = (size_t)blockIdx.x * blockDim.x + threadIdx.x;
    if (i >= total) return;
    int j = (int)(i % ld);
    float v = 0.f;
    if (j < Nreal) v = ((float)X[i] - mean[j]) * rstd[j] * gamma[j] + beta[j];
    X[i] = (_Float16)v;
}

// ---------------------------------------------------------------------------
// 6) fused BN + segment-max (order-preserving uint encode + atomicMax)
// ---------------------------------------------------------------------------
__global__ void psp_bn_segmax(const _Float16* __restrict__ F, int ld, int Nreal,
                              const int* __restrict__ si, int S,
                              const float* __restrict__ mean, const float* __restrict__ rstd,
                              const float* __restrict__ gamma, const float* __restrict__ beta,
                              unsigned* __restrict__ poolEnc, int poolLd) {
    size_t i = (size_t)blockIdx.x * blockDim.x + threadIdx.x;
    size_t total = (size_t)S * Nreal;
    if (i >= total) return;
    int s = (int)(i / Nreal), j = (int)(i % Nreal);
    float v = ((float)F[(size_t)s * ld + j] - mean[j]) * rstd[j] * gamma[j] + beta[j];
    unsigned u = __float_as_uint(v);
    unsigned enc = (u & 0x80000000u) ? ~u : (u | 0x80000000u);  // enc > 0 for all reals
    int seg = si[2 * s + 1];
    atomicMax(&poolEnc[(size_t)seg * poolLd + j], enc);
}

// ---------------------------------------------------------------------------
// 7) decode pooled maxima to f16 (sentinel 0 == empty segment == -inf -> 0)
// ---------------------------------------------------------------------------
__global__ void psp_pool_decode(const unsigned* __restrict__ poolEnc,
                                _Float16* __restrict__ pooled,
                                size_t total, int ld, int Nreal) {
    size_t i = (size_t)blockIdx.x * blockDim.x + threadIdx.x;
    if (i >= total) return;
    int j = (int)(i % ld);
    float v = 0.f;
    if (j < Nreal) {
        unsigned e = poolEnc[i];
        if (e != 0u) {
            unsigned u = (e & 0x80000000u) ? (e & 0x7FFFFFFFu) : ~e;
            v = __uint_as_float(u);
        }
    }
    pooled[i] = (_Float16)v;
}

// ---------------------------------------------------------------------------
// 8) final BN apply -> fp32 d_out (unpadded, M x Nreal contiguous)
// ---------------------------------------------------------------------------
__global__ void psp_bn_apply_final(const _Float16* __restrict__ X, int ld, int Nreal, size_t total,
                                   const float* __restrict__ mean, const float* __restrict__ rstd,
                                   const float* __restrict__ gamma, const float* __restrict__ beta,
                                   float* __restrict__ out) {
    size_t i = (size_t)blockIdx.x * blockDim.x + threadIdx.x;
    if (i >= total) return;
    int r = (int)(i / Nreal), j = (int)(i % Nreal);
    out[i] = ((float)X[(size_t)r * ld + j] - mean[j]) * rstd[j] * gamma[j] + beta[j];
}

// ---------------------------------------------------------------------------
// host launch
// ---------------------------------------------------------------------------
static inline size_t psp_align(size_t x) { return (x + 255) & ~(size_t)255; }

extern "C" void kernel_launch(void* const* d_in, const int* in_sizes, int n_in,
                              void* d_out, int out_size, void* d_ws, size_t ws_size,
                              hipStream_t stream) {
    const float* pc  = (const float*)d_in[0];
    const int*   kpi = (const int*)d_in[1];
    const int*   si  = (const int*)d_in[2];
    const int S = in_sizes[2] / 2;
    const int K = in_sizes[1];

    // params: point layers 0..3 then out layers 0..1, each {W,b,gamma,beta}
    const float* Wl[6]; const float* bl[6]; const float* gl[6]; const float* betal[6];
    for (int l = 0; l < 6; ++l) {
        Wl[l]    = (const float*)d_in[3 + 4 * l + 0];
        bl[l]    = (const float*)d_in[3 + 4 * l + 1];
        gl[l]    = (const float*)d_in[3 + 4 * l + 2];
        betal[l] = (const float*)d_in[3 + 4 * l + 3];
    }
    const int Kin[6]   = {3, 32, 64, 128, 300, 300};
    const int Nout[6]  = {32, 64, 128, 300, 300, 300};
    const int Kp[6]    = {32, 32, 64, 128, 320, 320};
    const int Np[6]    = {32, 64, 128, 320, 320, 320};
    const int Mrows[6] = {S, S, S, S, K, K};

    char* base = (char*)d_ws;
    size_t off = 0;
    auto carve = [&](size_t bytes) { char* p = base + off; off += psp_align(bytes); return p; };

    _Float16* act[7];
    act[0] = (_Float16*)carve((size_t)S * 32  * 2);
    act[1] = (_Float16*)carve((size_t)S * 32  * 2);
    act[2] = (_Float16*)carve((size_t)S * 64  * 2);
    act[3] = (_Float16*)carve((size_t)S * 128 * 2);
    act[4] = (_Float16*)carve((size_t)S * 320 * 2);
    act[5] = (_Float16*)carve((size_t)K * 320 * 2);
    act[6] = (_Float16*)carve((size_t)K * 320 * 2);
    unsigned*  poolEnc = (unsigned*)carve((size_t)K * 320 * 4);
    _Float16* pooled   = (_Float16*)carve((size_t)K * 320 * 2);
    _Float16* Wp[6]; float* bp[6];
    for (int l = 0; l < 6; ++l) {
        Wp[l] = (_Float16*)carve((size_t)Kp[l] * Np[l] * 2);
        bp[l] = (float*)carve((size_t)Np[l] * 4);
    }
    float* sum   = (float*)carve(320 * 4);
    float* sumsq = (float*)carve(320 * 4);
    float* mean  = (float*)carve(320 * 4);
    float* rstd  = (float*)carve(320 * 4);
    (void)ws_size; (void)n_in;

    const int TB = 256;

    auto run_stats = [&](const _Float16* X, int M, int ld, int Nreal) {
        hipMemsetAsync(sum,   0, 320 * 4, stream);
        hipMemsetAsync(sumsq, 0, 320 * 4, stream);
        size_t total = (size_t)M * ld;
        size_t want = (total + TB - 1) / TB;
        unsigned grid = (unsigned)(want < 4096 ? want : 4096);
        psp_bn_partial<<<grid, TB, 0, stream>>>(X, total, ld, sum, sumsq);
        psp_bn_finalize<<<(Nreal + TB - 1) / TB, TB, 0, stream>>>(sum, sumsq, M, Nreal, mean, rstd);
    };
    auto run_gemm = [&](const _Float16* A, int l, _Float16* O) {
        const int M = Mrows[l];
        const int Nt = Np[l] >> 4;
        if (Nt == 2) {   // layer 0: one group of 2 N-tiles
            const int waves = (M >> 4);
            psp_wmma_gemm_bias_relu<2><<<(waves + 7) / 8, TB, 0, stream>>>(
                A, Wp[l], bp[l], O, M, Kp[l], Np[l]);
        } else {         // Nt divisible by 4
            const int waves = (M >> 4) * (Nt / 4);
            psp_wmma_gemm_bias_relu<4><<<(waves + 7) / 8, TB, 0, stream>>>(
                A, Wp[l], bp[l], O, M, Kp[l], Np[l]);
        }
    };

    // 1) gather + diff
    psp_gather_diff<<<(S + TB - 1) / TB, TB, 0, stream>>>(pc, kpi, si, S, act[0]);

    // 2) pad + swizzle all weights
    for (int l = 0; l < 6; ++l) {
        int tot = Kp[l] * Np[l];
        psp_pad_weights_swz<<<(tot + TB - 1) / TB, TB, 0, stream>>>(
            Wl[l], bl[l], Kin[l], Nout[l], Kp[l], Np[l], Wp[l], bp[l]);
    }

    // point MLP: layers 0..3
    const _Float16* in = act[0];
    for (int l = 0; l < 4; ++l) {
        run_gemm(in, l, act[l + 1]);
        run_stats(act[l + 1], Mrows[l], Np[l], Nout[l]);
        if (l < 3) {
            size_t tot = (size_t)Mrows[l] * Np[l];
            psp_bn_apply<<<(unsigned)((tot + TB - 1) / TB), TB, 0, stream>>>(
                act[l + 1], tot, Np[l], Nout[l], mean, rstd, gl[l], betal[l]);
            in = act[l + 1];
        }
    }

    // segment max: fused BN(layer3) + encode + atomicMax, then decode
    hipMemsetAsync(poolEnc, 0, (size_t)K * 320 * 4, stream);
    {
        size_t tot = (size_t)S * 300;
        psp_bn_segmax<<<(unsigned)((tot + TB - 1) / TB), TB, 0, stream>>>(
            act[4], 320, 300, si, S, mean, rstd, gl[3], betal[3], poolEnc, 320);
        size_t ptot = (size_t)K * 320;
        psp_pool_decode<<<(unsigned)((ptot + TB - 1) / TB), TB, 0, stream>>>(
            poolEnc, pooled, ptot, 320, 300);
    }

    // out MLP: layers 4..5
    in = pooled;
    for (int l = 4; l < 6; ++l) {
        run_gemm(in, l, act[l + 1]);
        run_stats(act[l + 1], Mrows[l], Np[l], Nout[l]);
        if (l == 4) {
            size_t tot = (size_t)Mrows[l] * Np[l];
            psp_bn_apply<<<(unsigned)((tot + TB - 1) / TB), TB, 0, stream>>>(
                act[l + 1], tot, Np[l], Nout[l], mean, rstd, gl[l], betal[l]);
            in = act[l + 1];
        } else {
            size_t tot = (size_t)Mrows[l] * 300;
            psp_bn_apply_final<<<(unsigned)((tot + TB - 1) / TB), TB, 0, stream>>>(
                act[l + 1], Np[l], 300, tot, mean, rstd, gl[l], betal[l], (float*)d_out);
        }
    }
    (void)out_size;
}
